// QuantizedLinear_42984032698391
// MI455X (gfx1250) — matrix operand order repaired and checked by reading, not compiled, not run on hardware
//
#include <hip/hip_runtime.h>
#include <stdint.h>

// QuantizedLinear for MI455X (gfx1250, wave32, WMMA):
//   out[M,N] = (x[M,K] @ W[N,K]^T) * scale[N] + bias[N]
//   M = 8192, N = 4096, K = 4096
// f16 WMMA path (int8 exact in f16), fp32 accumulate, double-buffered LDS,
// magic-number int8->f16 unpack (v_perm_b32 + v_pk_add_f16).

typedef _Float16 v8h  __attribute__((ext_vector_type(8)));
typedef _Float16 v16h __attribute__((ext_vector_type(16)));
typedef _Float16 h2   __attribute__((ext_vector_type(2)));
typedef float    v8f  __attribute__((ext_vector_type(8)));

#define K_DIM 4096   // inC
#define N_DIM 4096   // outC
#define BM    128
#define BN    128
#define BK    32
#define LDA   40     // f16 per LDS row: 32 + 8 pad (80B stride, 16B aligned)
#define LDB   40
#define NITER (K_DIM / BK)

__device__ __forceinline__ v8h cvt_f32x8(float4 a, float4 b) {
    v8h r;
    r[0] = (_Float16)a.x; r[1] = (_Float16)a.y; r[2] = (_Float16)a.z; r[3] = (_Float16)a.w;
    r[4] = (_Float16)b.x; r[5] = (_Float16)b.y; r[6] = (_Float16)b.z; r[7] = (_Float16)b.w;
    return r;
}

// 4 signed bytes -> 4 exact f16 in two half2's.
// pack(0x64, b^0x80) is the f16 bit pattern of 1024 + (b+128); subtract 1152.
__device__ __forceinline__ void i8x4_to_f16(uint32_t d, h2& lo, h2& hi) {
    uint32_t u  = d ^ 0x80808080u;
    uint32_t pl = __builtin_amdgcn_perm(0x64646464u, u, 0x05010400u); // bytes [b0,64,b1,64]
    uint32_t ph = __builtin_amdgcn_perm(0x64646464u, u, 0x05030402u); // bytes [b2,64,b3,64]
    const _Float16 mg = (_Float16)1152.0f;
    lo = __builtin_bit_cast(h2, pl) - mg;   // v_pk_add_f16 (neg)
    hi = __builtin_bit_cast(h2, ph) - mg;
}

__device__ __forceinline__ v8h unpack_w8(uint32_t d0, uint32_t d1) {
    h2 a, b, c, d;
    i8x4_to_f16(d0, a, b);
    i8x4_to_f16(d1, c, d);
    v8h r;
    r[0] = a[0]; r[1] = a[1]; r[2] = b[0]; r[3] = b[1];
    r[4] = c[0]; r[5] = c[1]; r[6] = d[0]; r[7] = d[1];
    return r;
}

__global__ __launch_bounds__(256) void qlinear_wmma_f16(
    const float*  __restrict__ x,      // [M, K]
    const int8_t* __restrict__ w,      // [N, K]
    const float*  __restrict__ scale,  // [N]
    const float*  __restrict__ bias,   // [N]
    float*        __restrict__ out)    // [M, N]
{
    __shared__ _Float16 lds_a[2][BM * LDA];
    __shared__ _Float16 lds_b[2][BN * LDB];

    const int t      = threadIdx.x;
    const int lane   = t & 31;        // wave32
    const int wave   = t >> 5;        // 0..7
    const int wave_m = wave >> 1;     // 0..3 -> 32 rows each
    const int wave_n = wave & 1;      // 0..1 -> 64 cols each
    const int l16    = lane >> 4;     // half-wave id
    const int lmod   = lane & 15;

    const int block_n = blockIdx.x;   // fastest: x panel shared in L2
    const int block_m = blockIdx.y;

    // Cooperative loader: thread t -> row t/2, 16-element K segment (t%2)*16
    const int lrow = t >> 1;              // 0..127
    const int lseg = (t & 1) << 4;        // 0 or 16

    const float*  xg = x + (size_t)(block_m * BM + lrow) * K_DIM + lseg;
    const int8_t* wg = w + (size_t)(block_n * BN + lrow) * K_DIM + lseg;

    v8f acc[2][4] = {};   // 2 M-tiles x 4 N-tiles of 16x16 f32

    float4 xf0, xf1, xf2, xf3;
    int4   wb;

    // ---- preload + stage tile 0 ----
    {
        const float4* xp = (const float4*)xg;
        xf0 = xp[0]; xf1 = xp[1]; xf2 = xp[2]; xf3 = xp[3];
        wb = *(const int4*)wg;
        *(v8h*)&lds_a[0][lrow * LDA + lseg    ] = cvt_f32x8(xf0, xf1);
        *(v8h*)&lds_a[0][lrow * LDA + lseg + 8] = cvt_f32x8(xf2, xf3);
        *(v8h*)&lds_b[0][lrow * LDB + lseg    ] = unpack_w8((uint32_t)wb.x, (uint32_t)wb.y);
        *(v8h*)&lds_b[0][lrow * LDB + lseg + 8] = unpack_w8((uint32_t)wb.z, (uint32_t)wb.w);
    }
    __syncthreads();

    int buf = 0;
    for (int it = 0; it < NITER; ++it) {
        // ---- issue global loads for next tile (full tile of latency hiding) ----
        if (it + 1 < NITER) {
            const float4* xp = (const float4*)(xg + (it + 1) * BK);
            xf0 = xp[0]; xf1 = xp[1]; xf2 = xp[2]; xf3 = xp[3];
            wb = *(const int4*)(wg + (it + 1) * BK);
            if (it + 2 < NITER) {
                __builtin_prefetch(xg + (it + 2) * BK, 0, 0);   // global_prefetch_b8
                __builtin_prefetch(wg + (it + 2) * BK, 0, 0);
            }
        }

        // ---- fragments per ISA VGPR layout (CDNA5 ISA 7.12.2) ----
        const _Float16* A = lds_a[buf];
        const _Float16* B = lds_b[buf];
        v16h afrag[2];
        #pragma unroll
        for (int i = 0; i < 2; ++i) {
            // A 16x32: lanes 0-15 hold K 0-7 / 16-23, lanes 16-31 hold K 8-15 / 24-31
            const _Float16* p = &A[(wave_m * 32 + i * 16 + lmod) * LDA + l16 * 8];
            v8h lo = *(const v8h*)p;
            v8h hi = *(const v8h*)(p + 16);
            afrag[i] = __builtin_shufflevector(lo, hi, 0,1,2,3,4,5,6,7,8,9,10,11,12,13,14,15);
        }
        v16h bfrag[4];
        #pragma unroll
        for (int j = 0; j < 4; ++j) {
            // B 32x16: lane holds col N=lane%16, lanes 0-15 hold K 0-7 / 16-23, lanes 16-31 hold K 8-15 / 24-31
            const _Float16* p = &B[(wave_n * 64 + j * 16 + lmod) * LDB + l16 * 8];
            v8h lo = *(const v8h*)p;
            v8h hi = *(const v8h*)(p + 16);
            bfrag[j] = __builtin_shufflevector(lo, hi, 0,1,2,3,4,5,6,7,8,9,10,11,12,13,14,15);
        }

        #pragma unroll
        for (int i = 0; i < 2; ++i)
            #pragma unroll
            for (int j = 0; j < 4; ++j)
                acc[i][j] = __builtin_amdgcn_wmma_f32_16x16x32_f16(
                    false, afrag[i], false, bfrag[j],
                    (short)0, acc[i][j], false, false);

        // ---- stage next tile into the other buffer ----
        if (it + 1 < NITER) {
            _Float16* As = lds_a[buf ^ 1];
            _Float16* Bs = lds_b[buf ^ 1];
            *(v8h*)&As[lrow * LDA + lseg    ] = cvt_f32x8(xf0, xf1);
            *(v8h*)&As[lrow * LDA + lseg + 8] = cvt_f32x8(xf2, xf3);
            *(v8h*)&Bs[lrow * LDB + lseg    ] = unpack_w8((uint32_t)wb.x, (uint32_t)wb.y);
            *(v8h*)&Bs[lrow * LDB + lseg + 8] = unpack_w8((uint32_t)wb.z, (uint32_t)wb.w);
        }

        __syncthreads();   // single barrier per iteration (ping-pong buffers)
        buf ^= 1;
    }

    // ---- epilogue: out = acc * scale[n] + bias[n] ----
    // C/D layout: lanes 0-15 -> M = r, lanes 16-31 -> M = 8 + r, N = lane%16
    #pragma unroll
    for (int j = 0; j < 4; ++j) {
        const int n_g = block_n * BN + wave_n * 64 + j * 16 + lmod;
        const float s = scale[n_g];
        const float b = bias[n_g];
        #pragma unroll
        for (int i = 0; i < 2; ++i) {
            const int m_base = block_m * BM + wave_m * 32 + i * 16 + l16 * 8;
            float* op = out + (size_t)m_base * N_DIM + n_g;
            #pragma unroll
            for (int r = 0; r < 8; ++r)
                op[(size_t)r * N_DIM] = acc[i][j][r] * s + b;
        }
    }
}

extern "C" void kernel_launch(void* const* d_in, const int* in_sizes, int n_in,
                              void* d_out, int out_size, void* d_ws, size_t ws_size,
                              hipStream_t stream) {
    const float*  x     = (const float*)d_in[0];   // [4,2048,4096] fp32
    const int8_t* w     = (const int8_t*)d_in[1];  // [4096,4096] int8
    const float*  scale = (const float*)d_in[2];   // [4096]
    const float*  bias  = (const float*)d_in[3];   // [4096]
    float*        out   = (float*)d_out;           // [4,2048,4096] fp32

    const int M = in_sizes[0] / K_DIM;             // 8192
    dim3 grid(N_DIM / BN, M / BM);                 // (32, 64)
    qlinear_wmma_f16<<<grid, 256, 0, stream>>>(x, w, scale, bias, out);
}
